// CasualSelfAttention_7679401525489
// MI455X (gfx1250) — matrix operand order repaired and checked by reading, not compiled, not run on hardware
//
#include <hip/hip_runtime.h>
#include <hip/hip_bf16.h>

typedef __attribute__((ext_vector_type(16))) _Float16 v16h;
typedef __attribute__((ext_vector_type(8)))  _Float16 v8h;
typedef __attribute__((ext_vector_type(8)))  float    v8f;

#define BATCH   2
#define TSEQ    2048
#define N_EMBD  1024
#define N_HEAD  16
#define HEAD    64
#define MROWS   (BATCH * TSEQ)   // 4096

// ---------- WMMA helpers (CDNA5 wave32, v_wmma_f32_16x16x32_f16) ----------

__device__ __forceinline__ v8f wmma_f16(v16h a, v16h b, v8f c) {
  return __builtin_amdgcn_wmma_f32_16x16x32_f16(
      /*neg_a=*/false, a, /*neg_b=*/false, b,
      /*c_mod=*/(short)0, c, /*reuse_a=*/false, /*reuse_b=*/false);
}

// A fragment, 16x32 f16, row-major source, row stride ld (halfs).
// lane 0-15: M=lane, K chunks {0-7, 16-23}; lane 16-31: M=lane-16, K {8-15, 24-31}.
__device__ __forceinline__ v16h load_a_frag(const _Float16* __restrict__ base,
                                            int ld, int lane) {
  int m  = lane & 15;
  int hi = (lane >> 4) & 1;
  const _Float16* row = base + (size_t)m * ld + hi * 8;
  v8h lo  = *(const v8h*)(row);
  v8h hi8 = *(const v8h*)(row + 16);
  return __builtin_shufflevector(lo, hi8, 0,1,2,3,4,5,6,7,8,9,10,11,12,13,14,15);
}

// B fragment, 32x16 f16 (KxN), where source S is row-major with rows indexed by N
// (i.e. B[k][n] = S[n][k]), row stride ld. lane group 0 holds K {0-7, 16-23}, group 1 K {8-15, 24-31}.
__device__ __forceinline__ v16h load_b_frag(const _Float16* __restrict__ base,
                                            int ld, int lane) {
  int n  = lane & 15;
  int hi = (lane >> 4) & 1;
  const _Float16* row = base + (size_t)n * ld + hi * 8;
  v8h lo  = *(const v8h*)(row);
  v8h hi8 = *(const v8h*)(row + 16);
  return __builtin_shufflevector(lo, hi8, 0,1,2,3,4,5,6,7,8,9,10,11,12,13,14,15);
}

// ---------- 32x64 wave-tile GEMM body, K=N_EMBD, ping-pong pipelined ----------
// acc[rr][c] covers rows m0+rr*16.., cols n0+c*16.. ; arow0/arow1 point at the
// two 16-row A slabs, wbase at W row n0 (B operand: lane=N loads its W row).

__device__ __forceinline__ void gemm_tile_32x64(
    const _Float16* __restrict__ arow0, const _Float16* __restrict__ arow1,
    const _Float16* __restrict__ wbase, int lane, v8f acc[2][4]) {
  v16h a0A, a1A, bA[4], a0B, a1B, bB[4];

  // prologue: ping buffer <- k=0
  a0A = load_a_frag(arow0, N_EMBD, lane);
  a1A = load_a_frag(arow1, N_EMBD, lane);
#pragma unroll
  for (int c = 0; c < 4; ++c)
    bA[c] = load_b_frag(wbase + (size_t)(c * 16) * N_EMBD, N_EMBD, lane);

  for (int k = 0; k < N_EMBD; k += 64) {
    // load pong <- k+32 while consuming ping
    int k1 = k + 32;
    a0B = load_a_frag(arow0 + k1, N_EMBD, lane);
    a1B = load_a_frag(arow1 + k1, N_EMBD, lane);
#pragma unroll
    for (int c = 0; c < 4; ++c)
      bB[c] = load_b_frag(wbase + (size_t)(c * 16) * N_EMBD + k1, N_EMBD, lane);
#pragma unroll
    for (int c = 0; c < 4; ++c) acc[0][c] = wmma_f16(a0A, bA[c], acc[0][c]);
#pragma unroll
    for (int c = 0; c < 4; ++c) acc[1][c] = wmma_f16(a1A, bA[c], acc[1][c]);

    // load ping <- k+64 (clamped redundant reload on final step) while consuming pong
    int k2 = (k + 64 < N_EMBD) ? (k + 64) : k1;
    a0A = load_a_frag(arow0 + k2, N_EMBD, lane);
    a1A = load_a_frag(arow1 + k2, N_EMBD, lane);
#pragma unroll
    for (int c = 0; c < 4; ++c)
      bA[c] = load_b_frag(wbase + (size_t)(c * 16) * N_EMBD + k2, N_EMBD, lane);
#pragma unroll
    for (int c = 0; c < 4; ++c) acc[0][c] = wmma_f16(a0B, bB[c], acc[0][c]);
#pragma unroll
    for (int c = 0; c < 4; ++c) acc[1][c] = wmma_f16(a1B, bB[c], acc[1][c]);
  }
}

// ---------- fp32 -> f16 conversion ----------

__global__ void f32_to_f16_kernel(const float* __restrict__ src,
                                  _Float16* __restrict__ dst, int n) {
  int i = blockIdx.x * blockDim.x + threadIdx.x;
  if (i < n) dst[i] = (_Float16)src[i];
}

// ---------- stage 1: kqv = x @ W_kqv^T, scattered to q/k/vT per head ----------
// M=4096, N=3072, K=1024.  Wave tile: 32x64.

__global__ __launch_bounds__(128) void kqv_gemm_kernel(
    const _Float16* __restrict__ x16, const _Float16* __restrict__ w16,
    _Float16* __restrict__ q16, _Float16* __restrict__ k16,
    _Float16* __restrict__ vT16) {
  int lane = threadIdx.x & 31;
  int wave = threadIdx.x >> 5;
  int m0 = blockIdx.x * 32;
  int n0 = blockIdx.y * 256 + wave * 64;

  v8f acc[2][4] = {};
  gemm_tile_32x64(x16 + (size_t)m0 * N_EMBD,
                  x16 + (size_t)(m0 + 16) * N_EMBD,
                  w16 + (size_t)n0 * N_EMBD, lane, acc);

  int col = lane & 15;
  int hi  = (lane >> 4) & 1;
#pragma unroll
  for (int rr = 0; rr < 2; ++rr) {
#pragma unroll
    for (int c = 0; c < 4; ++c) {
#pragma unroll
      for (int r = 0; r < 8; ++r) {
        int mg = m0 + rr * 16 + r + 8 * hi;
        int bb = mg >> 11;             // / TSEQ
        int t  = mg & (TSEQ - 1);
        int ng = n0 + c * 16 + col;
        int sect = ng >> 10;           // 0=k, 1=q, 2=v (reference split order)
        int idx  = ng & (N_EMBD - 1);
        int h = idx >> 6, d = idx & (HEAD - 1);
        _Float16 v = (_Float16)acc[rr][c][r];
        size_t bh = (size_t)(bb * N_HEAD + h);
        if (sect == 0)      k16[(bh * TSEQ + t) * HEAD + d]  = v;
        else if (sect == 1) q16[(bh * TSEQ + t) * HEAD + d]  = v;
        else                vT16[(bh * HEAD + d) * TSEQ + t] = v;   // transposed V
      }
    }
  }
}

// ---------- stage 2: causal flash attention, one wave per 16-query tile ----------

__global__ __launch_bounds__(32) void flash_attn_kernel(
    const _Float16* __restrict__ q16, const _Float16* __restrict__ k16,
    const _Float16* __restrict__ vT16, _Float16* __restrict__ att16) {
  __shared__ __align__(16) _Float16 pt[16][32];   // P tile: C-layout -> A-layout bridge

  int lane = threadIdx.x & 31;
  int col  = lane & 15;
  int hi   = (lane >> 4) & 1;
  int qt = blockIdx.x;               // query tile, 0..T/16-1
  int bh = blockIdx.y;               // b*H + h
  int b = bh >> 4, h = bh & 15;
  int q0 = qt * 16;

  const _Float16* qbase = q16  + ((size_t)bh * TSEQ + q0) * HEAD;
  const _Float16* kbase = k16  + (size_t)bh * TSEQ * HEAD;
  const _Float16* vbase = vT16 + (size_t)bh * HEAD * TSEQ;

  v16h qa0 = load_a_frag(qbase + 0,  HEAD, lane);   // d 0..31
  v16h qa1 = load_a_frag(qbase + 32, HEAD, lane);   // d 32..63

  v8f o[4] = {};
  float mrow[8], lrow[8];
#pragma unroll
  for (int r = 0; r < 8; ++r) { mrow[r] = -1e30f; lrow[r] = 0.f; }

  int nkb = (q0 >> 5) + 1;           // 32-key blocks covering keys 0..q0+15
  for (int kb = 0; kb < nkb; ++kb) {
    int t0 = kb * 32;

    // K fragments first, then V fragments: the score WMMAs only need the K
    // loads retired, so the V loads stay in flight under the softmax VALU work.
    v16h bk[2][2];
#pragma unroll
    for (int g = 0; g < 2; ++g) {
      bk[g][0] = load_b_frag(kbase + (size_t)(t0 + g * 16) * HEAD + 0,  HEAD, lane);
      bk[g][1] = load_b_frag(kbase + (size_t)(t0 + g * 16) * HEAD + 32, HEAD, lane);
    }
    v16h bv[4];
#pragma unroll
    for (int c = 0; c < 4; ++c)
      bv[c] = load_b_frag(vbase + (size_t)(c * 16) * TSEQ + t0, TSEQ, lane);

    // prefetch next key block's K tile and V columns (global_prefetch_b8)
    if (kb + 1 < nkb) {
      __builtin_prefetch(kbase + (size_t)(t0 + 32 + lane) * HEAD, 0, 3);
      __builtin_prefetch(vbase + (size_t)lane * TSEQ + t0 + 32, 0, 3);
      __builtin_prefetch(vbase + (size_t)(32 + lane) * TSEQ + t0 + 32, 0, 3);
    }

    v8f s[2] = {};
#pragma unroll
    for (int g = 0; g < 2; ++g) {
      s[g] = wmma_f16(qa0, bk[g][0], s[g]);
      s[g] = wmma_f16(qa1, bk[g][1], s[g]);
    }

    bool diag = (kb == nkb - 1);     // only the diagonal block needs masking
#pragma unroll
    for (int g = 0; g < 2; ++g) {
#pragma unroll
      for (int r = 0; r < 8; ++r) {
        float v = s[g][r] * 0.125f;                       // D^-0.5 = 1/8
        int key = t0 + g * 16 + col;
        int qry = q0 + r + 8 * hi;
        if (diag && key > qry) v = -1e30f;
        s[g][r] = v;
      }
    }

    // online softmax: row stats replicated across each 16-lane group
    float corr[8];
#pragma unroll
    for (int r = 0; r < 8; ++r) {
      float vm = fmaxf(s[0][r], s[1][r]);
      vm = fmaxf(vm, __shfl_xor(vm, 1));
      vm = fmaxf(vm, __shfl_xor(vm, 2));
      vm = fmaxf(vm, __shfl_xor(vm, 4));
      vm = fmaxf(vm, __shfl_xor(vm, 8));
      float mnew = fmaxf(mrow[r], vm);
      corr[r] = __expf(mrow[r] - mnew);
      mrow[r] = mnew;
      float p0 = __expf(s[0][r] - mnew);
      float p1 = __expf(s[1][r] - mnew);
      s[0][r] = p0; s[1][r] = p1;
      float sum = p0 + p1;
      sum += __shfl_xor(sum, 1);
      sum += __shfl_xor(sum, 2);
      sum += __shfl_xor(sum, 4);
      sum += __shfl_xor(sum, 8);
      lrow[r] = lrow[r] * corr[r] + sum;
    }
#pragma unroll
    for (int c = 0; c < 4; ++c)
#pragma unroll
      for (int r = 0; r < 8; ++r) o[c][r] *= corr[r];

    // transpose P (C-layout) -> A-layout via LDS
#pragma unroll
    for (int r = 0; r < 8; ++r) {
      pt[r + 8 * hi][col]      = (_Float16)s[0][r];
      pt[r + 8 * hi][16 + col] = (_Float16)s[1][r];
    }
    __syncthreads();
    v8h plo = *(const v8h*)(&pt[col][hi * 8]);
    v8h phi = *(const v8h*)(&pt[col][16 + hi * 8]);
    v16h pa = __builtin_shufflevector(plo, phi,
                                      0,1,2,3,4,5,6,7,8,9,10,11,12,13,14,15);
    __syncthreads();

    // O += P @ V  (B operand from transposed V -> contiguous loads)
#pragma unroll
    for (int c = 0; c < 4; ++c) o[c] = wmma_f16(pa, bv[c], o[c]);
  }

  // epilogue: normalize and write to att16[b][t][h*64 + d]
#pragma unroll
  for (int r = 0; r < 8; ++r) {
    float inv = 1.0f / lrow[r];
    int t = q0 + r + 8 * hi;
#pragma unroll
    for (int c = 0; c < 4; ++c) {
      att16[((size_t)b * TSEQ + t) * N_EMBD + h * HEAD + c * 16 + col] =
          (_Float16)(o[c][r] * inv);
    }
  }
}

// ---------- stage 3: out = att @ W_proj^T + b_proj (fp32 out) ----------
// M=4096, N=1024.  Same 32x64 ping-pong wave tile.

__global__ __launch_bounds__(128) void proj_gemm_kernel(
    const _Float16* __restrict__ att16, const _Float16* __restrict__ w16,
    const float* __restrict__ bias, float* __restrict__ out) {
  int lane = threadIdx.x & 31;
  int wave = threadIdx.x >> 5;
  int m0 = blockIdx.x * 32;
  int n0 = blockIdx.y * 256 + wave * 64;

  v8f acc[2][4] = {};
  gemm_tile_32x64(att16 + (size_t)m0 * N_EMBD,
                  att16 + (size_t)(m0 + 16) * N_EMBD,
                  w16 + (size_t)n0 * N_EMBD, lane, acc);

  int col = lane & 15;
  int hi  = (lane >> 4) & 1;
#pragma unroll
  for (int rr = 0; rr < 2; ++rr) {
#pragma unroll
    for (int c = 0; c < 4; ++c) {
#pragma unroll
      for (int r = 0; r < 8; ++r) {
        int n = n0 + c * 16 + col;
        out[(size_t)(m0 + rr * 16 + r + 8 * hi) * N_EMBD + n] =
            acc[rr][c][r] + bias[n];
      }
    }
  }
}

// ---------- launch ----------

extern "C" void kernel_launch(void* const* d_in, const int* in_sizes, int n_in,
                              void* d_out, int out_size, void* d_ws, size_t ws_size,
                              hipStream_t stream) {
  const float* x      = (const float*)d_in[0];   // [2,2048,1024]
  const float* W_kqv  = (const float*)d_in[1];   // [3072,1024]
  const float* W_proj = (const float*)d_in[2];   // [1024,1024]
  const float* b_proj = (const float*)d_in[3];   // [1024]
  float* out = (float*)d_out;                    // [2,2048,1024] fp32

  char* ws = (char*)d_ws;
  size_t off = 0;
  auto take = [&](size_t bytes) -> char* {
    char* p = ws + off;
    off += (bytes + 255) & ~(size_t)255;
    return p;
  };
  const int nX = MROWS * N_EMBD;            // 4,194,304
  const int nWk = 3 * N_EMBD * N_EMBD;      // 3,145,728
  const int nWp = N_EMBD * N_EMBD;          // 1,048,576

  _Float16* x16    = (_Float16*)take((size_t)nX  * 2);
  _Float16* wkqv16 = (_Float16*)take((size_t)nWk * 2);
  _Float16* wproj16= (_Float16*)take((size_t)nWp * 2);
  _Float16* q16    = (_Float16*)take((size_t)nX  * 2);
  _Float16* k16    = (_Float16*)take((size_t)nX  * 2);
  _Float16* vT16   = (_Float16*)take((size_t)nX  * 2);
  _Float16* att16  = (_Float16*)take((size_t)nX  * 2);

  f32_to_f16_kernel<<<(nX  + 255) / 256, 256, 0, stream>>>(x, x16, nX);
  f32_to_f16_kernel<<<(nWk + 255) / 256, 256, 0, stream>>>(W_kqv, wkqv16, nWk);
  f32_to_f16_kernel<<<(nWp + 255) / 256, 256, 0, stream>>>(W_proj, wproj16, nWp);

  // kqv: M=4096 (128 tiles of 32), N=3072 (12 blocks of 256)
  kqv_gemm_kernel<<<dim3(MROWS / 32, 12), 128, 0, stream>>>(x16, wkqv16,
                                                            q16, k16, vT16);
  // attention: T/16 = 128 query tiles, B*H = 32 head slices
  flash_attn_kernel<<<dim3(TSEQ / 16, BATCH * N_HEAD), 32, 0, stream>>>(
      q16, k16, vT16, att16);
  // proj: M=4096 (128 tiles of 32), N=1024 (4 blocks of 256)
  proj_gemm_kernel<<<dim3(MROWS / 32, 4), 128, 0, stream>>>(att16, wproj16,
                                                            b_proj, out);
}
